// RnRoPE_76330158785253
// MI455X (gfx1250) — matrix operand order repaired and checked by reading, not compiled, not run on hardware
//
#include <hip/hip_runtime.h>
#include <stdint.h>

// ---------------- problem constants ----------------
constexpr int D    = 128;
constexpr int Bb   = 4;
constexpr int Hh   = 16;
constexpr int S    = 4096;
constexpr int ROWS = Bb * Hh * S;          // 262144 rows per matrix
constexpr int NT   = 8;                    // 128 / 16 output column tiles
constexpr int KC   = 4;                    // 128 / 32 k chunks

// LDS layout strides (chosen for bank-conflict-free gathers)
constexpr int TS    = 134;                 // bf16 table row stride (elements)
constexpr int TABSZ = 128 * TS;            // elements per table
constexpr int TAB_BYTES = 4 * TABSZ * 2;   // Th, Tl, Gh, Gl = 137216 (divisible by 8)
constexpr int SS    = 132;                 // f32 stage row stride (floats)
constexpr int STAGE_FLOATS = 16 * SS;      // per-wave stage (16 rows)
constexpr int STAGE_BYTES  = STAGE_FLOATS * 4;
constexpr int WAVES_PER_BLOCK = 16;        // 512 threads
constexpr int SMEM_MAIN = TAB_BYTES + WAVES_PER_BLOCK * STAGE_BYTES; // 272384 <= 320KB
constexpr int SMEM_BUILD = 128 * 128 * 4;  // 64KB Q matrix

typedef __attribute__((ext_vector_type(16))) __bf16 v16bf;
typedef __attribute__((ext_vector_type(8)))  float  v8f;
typedef __attribute__((ext_vector_type(4)))  unsigned int uint4v;
typedef __attribute__((ext_vector_type(8)))  int    int8v;
typedef __attribute__((ext_vector_type(4)))  int    int4v;

union BFrag {
    v16bf v;
    unsigned short u[16];
    unsigned int   d[8];
};

__device__ __forceinline__ unsigned short f32_to_bf16_rne(float x) {
    unsigned u = __builtin_bit_cast(unsigned, x);
    u += 0x7FFFu + ((u >> 16) & 1u);
    return (unsigned short)(u >> 16);
}
__device__ __forceinline__ float bf16_bits_to_f32(unsigned short h) {
    return __builtin_bit_cast(float, ((unsigned)h) << 16);
}
// Truncating hi/lo split: hi = trunc_bf16(x) (dropped bits land exactly in lo),
// lo = trunc_bf16(x - hi).  Residual error ~2^-16 relative — fp32-class once
// combined with the RNE-split Q tables via 3 WMMA products.
__device__ __forceinline__ void split_trunc(float x, unsigned short& h, unsigned short& l) {
    unsigned u = __builtin_bit_cast(unsigned, x);
    h = (unsigned short)(u >> 16);
    float hf = __builtin_bit_cast(float, u & 0xffff0000u);
    float lf = x - hf;
    l = (unsigned short)(__builtin_bit_cast(unsigned, lf) >> 16);
}

// ---------------------------------------------------------------------------
// Kernel 1: build Q = H63..H0 * I via Householder reflections, then emit
// bf16 hi/lo tables into workspace:
//   T[n][k] = Q[n][k]  (B-operand for y = x @ Q^T : B[k][n] = Q[n][k])
//   G[n][k] = Q[k][n]  (B-operand for z = r @ Q   : B[k][n] = Q[k][n])
// One thread owns one column of Q -> no barriers in the reflection chain.
// ---------------------------------------------------------------------------
extern __shared__ char dyn_smem[];

__global__ void build_q_tables(const float* __restrict__ vs,
                               unsigned short* __restrict__ tabs) {
    float* Qf = (float*)dyn_smem;          // 128x128
    const int j = threadIdx.x;             // blockDim.x == 128
    for (int i = 0; i < 128; ++i) Qf[i * 128 + j] = (i == j) ? 1.0f : 0.0f;

    for (int refl = 0; refl < 64; ++refl) {
        const float* v = vs + refl * 128;
        float vn = 1e-8f;                  // EPS, matches reference
        float w  = 0.0f;
        for (int i = 0; i < 128; ++i) {
            float vi = v[i];
            vn += vi * vi;
            w  += vi * Qf[i * 128 + j];
        }
        float coef = 2.0f / vn * w;
        for (int i = 0; i < 128; ++i) Qf[i * 128 + j] -= coef * v[i];
    }
    __syncthreads();                       // table emit reads other columns

    for (int t = threadIdx.x; t < 128 * 128; t += blockDim.x) {
        int n = t >> 7, kk = t & 127;
        float qnk = Qf[n * 128 + kk];
        float qkn = Qf[kk * 128 + n];
        unsigned short h, l;
        h = f32_to_bf16_rne(qnk);
        l = f32_to_bf16_rne(qnk - bf16_bits_to_f32(h));
        tabs[0 * TABSZ + n * TS + kk] = h;
        tabs[1 * TABSZ + n * TS + kk] = l;
        h = f32_to_bf16_rne(qkn);
        l = f32_to_bf16_rne(qkn - bf16_bits_to_f32(h));
        tabs[2 * TABSZ + n * TS + kk] = h;
        tabs[3 * TABSZ + n * TS + kk] = l;
    }
}

// ---------------------------------------------------------------------------
// Main kernel helpers
// ---------------------------------------------------------------------------

// Gather a 16x128 fp32 tile from the per-wave LDS stage into 4 (k-chunk)
// A-operand fragments in the ISA 16-bit A 16x32 layout, splitting each value
// into bf16 hi + lo on the fly.
//   lane L: m = L&15, half = L>>4; VGPR r holds k = (r<4?0:16)+half*8+(r&3)*2 (+1)
__device__ __forceinline__ void gather_a(const float* __restrict__ stage,
                                         int nl, int half,
                                         BFrag* __restrict__ ah,
                                         BFrag* __restrict__ al) {
#pragma unroll
    for (int kc = 0; kc < KC; ++kc) {
#pragma unroll
        for (int r = 0; r < 8; ++r) {
            int kk = kc * 32 + ((r & 4) ? 16 : 0) + half * 8 + (r & 3) * 2;
            const float* p = stage + nl * SS + kk;
            float x0 = p[0], x1 = p[1];
            unsigned short h0, l0, h1, l1;
            split_trunc(x0, h0, l0);
            split_trunc(x1, h1, l1);
            ah[kc].u[2 * r]     = h0;
            ah[kc].u[2 * r + 1] = h1;
            al[kc].u[2 * r]     = l0;
            al[kc].u[2 * r + 1] = l1;
        }
    }
}

// Load one (nt,kc) B-fragment pair from the LDS tables in the ISA 16-bit
// B 32x16 layout: lane L: n = L&15, half = L>>4; VGPR r holds k = (r<4?0:16)+half*8+(r&3)*2.
__device__ __forceinline__ void load_b(const unsigned short* __restrict__ tabh,
                                       const unsigned short* __restrict__ tabl,
                                       int nl, int half, int step,
                                       BFrag& bh, BFrag& bl) {
    const int nt = step >> 2, kc = step & 3;
    const unsigned short* rh = tabh + (nt * 16 + nl) * TS + half * 8 + kc * 32;
    const unsigned short* rl = tabl + (nt * 16 + nl) * TS + half * 8 + kc * 32;
#pragma unroll
    for (int r = 0; r < 8; ++r) {
        const int o = ((r & 4) ? 8 : 0) + 2 * r;
        bh.u[2 * r]     = rh[o];
        bh.u[2 * r + 1] = rh[o + 1];
        bl.u[2 * r]     = rl[o];
        bl.u[2 * r + 1] = rl[o + 1];
    }
}

// One 16x128 = (16x128)x(128x128) GEMM with split-bf16 error compensation:
//   acc = ah*Bh + al*Bh + ah*Bl   (f32 accumulate)
// Flattened 32-step pipeline with 4 B-fragment buffers and prefetch
// distance 2: each step's WMMAs run while the loads issued two steps ago are
// still young, so the s_wait_dscnt before each WMMA group tolerates ~8
// outstanding DS ops instead of forcing dscnt==0 (round 4 disasm).  The
// per-step sched_barrier(0) caps lookahead so pressure stays < 256 VGPRs
// (round 3 spilled without it).
__device__ __forceinline__ void gemm_tile(const unsigned short* __restrict__ tabh,
                                          const unsigned short* __restrict__ tabl,
                                          const BFrag* __restrict__ ah,
                                          const BFrag* __restrict__ al,
                                          int nl, int half,
                                          v8f* __restrict__ acc) {
    BFrag bh[4], bl[4];
    load_b(tabh, tabl, nl, half, 0, bh[0], bl[0]);
    load_b(tabh, tabl, nl, half, 1, bh[1], bl[1]);
    v8f c = {0.f, 0.f, 0.f, 0.f, 0.f, 0.f, 0.f, 0.f};
#pragma unroll
    for (int step = 0; step < NT * KC; ++step) {
        const int kc  = step & 3;
        const int cur = step & 3;              // buffer index = step % 4
        if (step + 2 < NT * KC)
            load_b(tabh, tabl, nl, half, step + 2,
                   bh[(step + 2) & 3], bl[(step + 2) & 3]);
        c = __builtin_amdgcn_wmma_f32_16x16x32_bf16(false, ah[kc].v, false, bh[cur].v,
                                                    (short)0, c, false, false);
        c = __builtin_amdgcn_wmma_f32_16x16x32_bf16(false, al[kc].v, false, bh[cur].v,
                                                    (short)0, c, false, false);
        c = __builtin_amdgcn_wmma_f32_16x16x32_bf16(false, ah[kc].v, false, bl[cur].v,
                                                    (short)0, c, false, false);
        if (kc == 3) {
            acc[step >> 2] = c;
            c = (v8f){0.f, 0.f, 0.f, 0.f, 0.f, 0.f, 0.f, 0.f};
        }
        __builtin_amdgcn_sched_barrier(0);
    }
}

// ---------------------------------------------------------------------------
// Kernel 2: TDM-loads the Q tables into LDS once per block, then per-wave
// 16-row tiles: async stage -> GEMM1 (x@Q^T) -> RoPE on accumulator
// fragments -> LDS transpose-stage -> GEMM2 (r@Q) -> store.
// ---------------------------------------------------------------------------
__global__ void __launch_bounds__(512, 1)
rnrope_main(const float* __restrict__ qin, const float* __restrict__ kin,
            const float* __restrict__ cosp, const float* __restrict__ sinp,
            const unsigned short* __restrict__ tabs,
            float* __restrict__ outq, float* __restrict__ outk) {
    unsigned short* tlds = (unsigned short*)dyn_smem;          // 4 tables
    float* stage_all     = (float*)(dyn_smem + TAB_BYTES);     // 16 waves x 16xSS

    const int tid  = threadIdx.x;
    const int lane = tid & 31;
    const int wave = tid >> 5;
    const int nl   = lane & 15;
    const int half = lane >> 4;
    const unsigned ldsDynBase = __builtin_amdgcn_groupstaticsize(); // dyn-LDS byte base

    // ---- Tensor Data Mover: pull all 4 bf16 tables (137216 B) into LDS.
    // 1D tile of 17152 x 8-byte elements (data_size=3 keeps tile_dim0 < 2^16).
    // TDM ignores EXEC, so issue from wave 0 only via a wave-uniform branch.
    if (tid < 32) {
        const unsigned NELEM = TAB_BYTES / 8;                  // 17152
        uint64_t ga = (uint64_t)(uintptr_t)tabs;
        uint4v g0;
        g0[0] = 1u;                                            // count=1, user desc
        g0[1] = ldsDynBase;                                    // lds_addr (bytes)
        g0[2] = (unsigned)(ga & 0xffffffffu);                  // global_addr[31:0]
        g0[3] = (unsigned)((ga >> 32) & 0x01ffffffu)           // global_addr[56:32]
              | (2u << 30);                                    // type=2 ("image")
        int8v g1;
        g1[0] = (int)(3u << 16);                               // wg_mask=0, data_size=3 (8B)
        g1[1] = (int)((NELEM & 0xffffu) << 16);                // tensor_dim0[15:0] @bits63:48
        g1[2] = (int)(1u << 16);                               // dim0[31:16]=0, tensor_dim1=1
        g1[3] = (int)(NELEM << 16);                            // dim1 hi=0, tile_dim0=17152
        g1[4] = 1;                                             // tile_dim1=1, tile_dim2=0
        g1[5] = (int)NELEM;                                    // tensor_dim0_stride[31:0]
        g1[6] = 0;                                             // stride0 hi, stride1 lo
        g1[7] = 0;
        int4v g2 = {0, 0, 0, 0};
        int4v g3 = {0, 0, 0, 0};
        int8v g4 = {0, 0, 0, 0, 0, 0, 0, 0};                   // unused trailing group
        __builtin_amdgcn_tensor_load_to_lds(g0, g1, g2, g3, g4, 0);
        __builtin_amdgcn_s_wait_tensorcnt(0);
    }
    __syncthreads();

    const unsigned short* Th = tlds + 0 * TABSZ;
    const unsigned short* Tl = tlds + 1 * TABSZ;
    const unsigned short* Gh = tlds + 2 * TABSZ;
    const unsigned short* Gl = tlds + 3 * TABSZ;
    float* stage = stage_all + wave * STAGE_FLOATS;
    const unsigned stageLdsByte = ldsDynBase + TAB_BYTES + wave * STAGE_BYTES;

    const int waveGlobal = blockIdx.x * WAVES_PER_BLOCK + wave;
    const int nWaves     = gridDim.x * WAVES_PER_BLOCK;
    const int nTiles     = ROWS / 16;

    for (int tile = waveGlobal; tile < nTiles; tile += nWaves) {
        const int row0 = tile * 16;
        const int b    = row0 / (Hh * S);
        const int s0   = row0 % S;                 // 16 | S, tile stays in one (b,h)
        const size_t csBase = ((size_t)(b * S + s0)) * D;

        for (int which = 0; which < 2; ++which) {
            const float* X = (which == 0 ? qin : kin) + (size_t)row0 * D;
            float*       Z = (which == 0 ? outq : outk) + (size_t)row0 * D;

            // ---- async-stage 16x128 fp32 tile straight into LDS (b128, GVS) ----
#pragma unroll
            for (int it = 0; it < 16; ++it) {
                const int t  = it * 32 + lane;
                const unsigned m  = (unsigned)t >> 5;
                const unsigned d4 = (unsigned)t & 31u;
                const unsigned ldsOff = stageLdsByte + m * (SS * 4) + d4 * 16u;
                const unsigned gOff   = (unsigned)t * 16u;
                asm volatile("global_load_async_to_lds_b128 %0, %1, %2"
                             :: "v"(ldsOff), "v"(gOff), "s"(X) : "memory");
            }
            asm volatile("s_wait_asynccnt 0x0" ::: "memory");

            // ---- GEMM1: y = x @ Q^T ----
            BFrag ah[KC], al[KC];
            gather_a(stage, nl, half, ah, al);
            __builtin_amdgcn_sched_barrier(0);
            v8f acc[NT];
            gemm_tile(Th, Tl, ah, al, nl, half, acc);

            // ---- RoPE on accumulator fragments ----
            // C/D layout: lane (nl,half), VGPR v -> (m = v + half*8, d = nt*16+nl)
            // rotate_half pairs d with d+-64 -> acc tile nt with nt+-4, same lane/VGPR.
#pragma unroll
            for (int nt = 0; nt < 4; ++nt) {
                const int d0 = nt * 16 + nl;
                const int d1 = d0 + 64;
#pragma unroll
                for (int v = 0; v < 8; ++v) {
                    const int m = v + half * 8;
                    const size_t off = csBase + (size_t)m * D;
                    float c0 = cosp[off + d0], sv0 = sinp[off + d0];
                    float c1 = cosp[off + d1], sv1 = sinp[off + d1];
                    float y0 = acc[nt][v], y1 = acc[nt + 4][v];
                    acc[nt][v]     = y0 * c0 - y1 * sv0;   // i < 64: y*c - y[i+64]*s
                    acc[nt + 4][v] = y1 * c1 + y0 * sv1;   // i >= 64: y*c + y[i-64]*s
                }
            }
            __builtin_amdgcn_sched_barrier(0);

            // ---- transpose-stage r through LDS (C layout -> A layout) ----
#pragma unroll
            for (int nt = 0; nt < NT; ++nt)
#pragma unroll
                for (int v = 0; v < 8; ++v) {
                    const int m = v + half * 8;
                    stage[m * SS + nt * 16 + nl] = acc[nt][v];
                }
            asm volatile("s_wait_dscnt 0x0" ::: "memory");   // per-wave LDS RAW

            // ---- GEMM2: z = r @ Q ----
            gather_a(stage, nl, half, ah, al);
            __builtin_amdgcn_sched_barrier(0);
            gemm_tile(Gh, Gl, ah, al, nl, half, acc);

            // ---- store (lanes 0..15 cover 16 consecutive floats) ----
#pragma unroll
            for (int nt = 0; nt < NT; ++nt)
#pragma unroll
                for (int v = 0; v < 8; ++v) {
                    const int m = v + half * 8;
                    Z[(size_t)m * D + nt * 16 + nl] = acc[nt][v];
                }
        }
    }
}

// ---------------------------------------------------------------------------
// Host launcher
// ---------------------------------------------------------------------------
extern "C" void kernel_launch(void* const* d_in, const int* in_sizes, int n_in,
                              void* d_out, int out_size, void* d_ws, size_t ws_size,
                              hipStream_t stream) {
    const float* q    = (const float*)d_in[0];
    const float* k    = (const float*)d_in[1];
    const float* cosp = (const float*)d_in[2];
    const float* sinp = (const float*)d_in[3];
    const float* vs   = (const float*)d_in[4];
    float* outq = (float*)d_out;
    float* outk = outq + (size_t)ROWS * D;
    unsigned short* tabs = (unsigned short*)d_ws;   // 137216 bytes used

    (void)in_sizes; (void)n_in; (void)out_size; (void)ws_size;

    (void)hipFuncSetAttribute((const void*)build_q_tables,
                              hipFuncAttributeMaxDynamicSharedMemorySize, SMEM_BUILD);
    (void)hipFuncSetAttribute((const void*)rnrope_main,
                              hipFuncAttributeMaxDynamicSharedMemorySize, SMEM_MAIN);

    build_q_tables<<<1, 128, SMEM_BUILD, stream>>>(vs, tabs);
    rnrope_main<<<256, 512, SMEM_MAIN, stream>>>(q, k, cosp, sinp, tabs, outq, outk);
}
